// EnsembleSurrogate_944892805218
// MI455X (gfx1250) — matrix operand order repaired and checked by reading, not compiled, not run on hardware
//
#include <hip/hip_runtime.h>
#include <hip/hip_bf16.h>

typedef __attribute__((ext_vector_type(16))) _Float16 v16h;
typedef __attribute__((ext_vector_type(8)))  float    v8f;
typedef __attribute__((ext_vector_type(4)))  unsigned int uv4;
typedef int v4i __attribute__((vector_size(16)));   // pointee type for async-LDS builtin

#define N_MODELS 16
#define IN_DIM   32
#define HID      64
#define OUT_DIM  8
#define WAVES_PER_BLOCK 8
#define ROWS_PER_WAVE   32                     // two 16-row WMMA sub-tiles per wave
#define ROWS_PER_BLOCK  (WAVES_PER_BLOCK * ROWS_PER_WAVE)   // 256

// Packed per-model weight blob: f16 transposed weights + f32 biases,
// laid out exactly as the LDS image so staging is a raw 16B-chunk copy.
#define OFF_W1 0                               // [64][32] f16
#define OFF_W2 2048                            // [64][64] f16
#define OFF_W3 6144                            // [64][64] f16
#define OFF_W4 10240                           // [16][64] f16 (N padded 8->16, zeros)
#define W_HALVES 11264
#define BIAS_OFF_BYTES (W_HALVES * 2)          // 22528
#define BIAS_FLOATS 208                        // b1(64) b2(64) b3(64) b4(16, padded)
#define BLOB_BYTES (BIAS_OFF_BYTES + BIAS_FLOATS * 4)       // 23360 (16B multiple)

#if __has_builtin(__builtin_amdgcn_global_load_async_to_lds_b128) && \
    __has_builtin(__builtin_amdgcn_s_wait_asynccnt)
#define HAVE_ASYNC_LDS 1
#else
#define HAVE_ASYNC_LDS 0
#endif

union FragU { v16h h; uv4 u[2]; };

// A-fragment (16x32 f16): lanes 0-15 hold K=0..7 / K=16..23,
// lanes 16-31 hold K=8..15 / K=24..31 (two b128 LDS loads).
__device__ __forceinline__ v16h load_fragA(const _Float16* rowp, int K0, int half) {
    FragU f;
    f.u[0] = *(const uv4*)(rowp + K0 + half * 8);
    f.u[1] = *(const uv4*)(rowp + K0 + half * 8 + 16);
    return f.h;
}

// B-fragment (32x16 f16): lanes 0-15 hold K=0..7 / K=16..23, lanes 16-31 hold K=8..15 / K=24..31.
// Weights are pre-transposed (Wt[n][k]) so this is two contiguous b128 loads.
__device__ __forceinline__ v16h load_fragB(const _Float16* Wt, int ld, int K0,
                                           int t, int r, int half) {
    const _Float16* p = Wt + (t * 16 + r) * ld + K0 + half * 8;
    FragU f;
    f.u[0] = *(const uv4*)(p);
    f.u[1] = *(const uv4*)(p + 16);
    return f.h;
}

__device__ __forceinline__ v8f wmma_f16(v16h a, v16h b, v8f c) {
    return __builtin_amdgcn_wmma_f32_16x16x32_f16(false, a, false, b,
                                                  (short)0, c, false, false);
}

__device__ __forceinline__ v8f v8f_splat(float v) {
    v8f r;
#pragma unroll
    for (int i = 0; i < 8; ++i) r[i] = v;
    return r;
}

// ReLU + f32->f16 + D-layout -> row-major LDS (feeds next layer's A-fragments).
__device__ __forceinline__ void relu_store(_Float16* hbuf, v8f acc,
                                           int base_row, int col, int half) {
#pragma unroll
    for (int v = 0; v < 8; ++v) {
        float h = __builtin_fmaxf(acc[v], 0.f);
        hbuf[(base_row + v + 8 * half) * HID + col] = (_Float16)h;
    }
}

// One 64->64 hidden layer for both 16-row sub-tiles (dual WMMA chains share B).
__device__ __forceinline__ void layer_hid(const _Float16* Wt, const float* bias,
                                          _Float16* hbuf, int r, int half,
                                          v16h a00, v16h a01, v16h a10, v16h a11) {
#pragma unroll
    for (int t = 0; t < 4; ++t) {
        v16h b0 = load_fragB(Wt, HID, 0,  t, r, half);
        v16h b1 = load_fragB(Wt, HID, 32, t, r, half);
        float bv = bias[t * 16 + r];
        v8f acc0 = v8f_splat(bv), acc1 = v8f_splat(bv);
        acc0 = wmma_f16(a00, b0, acc0);
        acc1 = wmma_f16(a10, b0, acc1);   // independent chain hides WMMA hazards
        acc0 = wmma_f16(a01, b1, acc0);
        acc1 = wmma_f16(a11, b1, acc1);
        relu_store(hbuf, acc0, 0,  t * 16 + r, half);
        relu_store(hbuf, acc1, 16, t * 16 + r, half);
    }
}

template <bool PACKED>
__global__ __launch_bounds__(256)
void ensemble_mlp_kernel(const float* __restrict__ x,
                         const float* __restrict__ W1, const float* __restrict__ b1,
                         const float* __restrict__ W2, const float* __restrict__ b2,
                         const float* __restrict__ W3, const float* __restrict__ b3,
                         const float* __restrict__ W4, const float* __restrict__ b4,
                         const unsigned char* __restrict__ wblob,
                         float* __restrict__ out, int batch) {
    __shared__ __align__(16) unsigned char smem[BLOB_BYTES];                // 23.4 KB
    __shared__ _Float16 sH[WAVES_PER_BLOCK][ROWS_PER_WAVE * HID];           // 32 KB

    _Float16* sW  = (_Float16*)smem;
    _Float16* sW1 = sW + OFF_W1;
    _Float16* sW2 = sW + OFF_W2;
    _Float16* sW3 = sW + OFF_W3;
    _Float16* sW4 = sW + OFF_W4;
    float* sb  = (float*)(smem + BIAS_OFF_BYTES);
    float* sb1 = sb;
    float* sb2 = sb + 64;
    float* sb3 = sb + 128;
    float* sb4 = sb + 192;

    const int tid  = threadIdx.x;
    const int wave = tid >> 5;
    const int lane = tid & 31;
    const int r    = lane & 15;
    const int half = lane >> 4;
    const int rowbase = blockIdx.x * ROWS_PER_BLOCK + wave * ROWS_PER_WAVE;

    // ---- x tiles -> two f16 A-fragments, kept in VGPRs for all models ----
    v16h xa0, xa1;
#pragma unroll
    for (int s = 0; s < 2; ++s) {
        const float* xr = x + (size_t)(rowbase + s * 16 + r) * IN_DIM + half * 8;
        float4 f0 = *(const float4*)(xr);
        float4 f1 = *(const float4*)(xr + 4);
        float4 f2 = *(const float4*)(xr + 16);
        float4 f3 = *(const float4*)(xr + 20);
        v16h a;
        a[0] = (_Float16)f0.x;  a[1] = (_Float16)f0.y;  a[2] = (_Float16)f0.z;  a[3] = (_Float16)f0.w;
        a[4] = (_Float16)f1.x;  a[5] = (_Float16)f1.y;  a[6] = (_Float16)f1.z;  a[7] = (_Float16)f1.w;
        a[8] = (_Float16)f2.x;  a[9] = (_Float16)f2.y;  a[10] = (_Float16)f2.z; a[11] = (_Float16)f2.w;
        a[12] = (_Float16)f3.x; a[13] = (_Float16)f3.y; a[14] = (_Float16)f3.z; a[15] = (_Float16)f3.w;
        if (s == 0) xa0 = a; else xa1 = a;
    }

    v8f psum0 = v8f_splat(0.f), psq0 = v8f_splat(0.f);
    v8f psum1 = v8f_splat(0.f), psq1 = v8f_splat(0.f);

    _Float16* hbuf = sH[wave];

    for (int m = 0; m < N_MODELS; ++m) {
        __syncthreads();  // previous model's compute done before restaging

        if (PACKED) {
            const unsigned char* g = wblob + (size_t)m * BLOB_BYTES;
#if HAVE_ASYNC_LDS
            for (int e = tid; e < BLOB_BYTES / 16; e += 256)
                __builtin_amdgcn_global_load_async_to_lds_b128(
                    (__attribute__((address_space(1))) v4i*)(g + e * 16),
                    (__attribute__((address_space(3))) v4i*)(smem + e * 16),
                    0, 0);
            __builtin_amdgcn_s_wait_asynccnt(0);
#else
            for (int e = tid; e < BLOB_BYTES / 16; e += 256)
                ((uv4*)smem)[e] = ((const uv4*)g)[e];
#endif
            if (tid == 0 && m + 1 < N_MODELS)
                __builtin_prefetch(wblob + (size_t)(m + 1) * BLOB_BYTES, 0, 1);
        } else {
            // Fallback: transpose+convert in-kernel (used only if d_ws too small).
            const float* g = W1 + (size_t)m * IN_DIM * HID;
            for (int e = tid; e < IN_DIM * HID; e += 256) {
                int k = e / HID, n = e % HID;
                sW1[n * IN_DIM + k] = (_Float16)g[e];
            }
            g = W2 + (size_t)m * HID * HID;
            for (int e = tid; e < HID * HID; e += 256) {
                int k = e / HID, n = e % HID;
                sW2[n * HID + k] = (_Float16)g[e];
            }
            g = W3 + (size_t)m * HID * HID;
            for (int e = tid; e < HID * HID; e += 256) {
                int k = e / HID, n = e % HID;
                sW3[n * HID + k] = (_Float16)g[e];
            }
            g = W4 + (size_t)m * HID * OUT_DIM;
            for (int e = tid; e < 16 * HID; e += 256) {
                int n = e >> 6, k = e & 63;
                sW4[e] = (n < OUT_DIM) ? (_Float16)g[k * OUT_DIM + n] : (_Float16)0.f;
            }
            if (tid < HID) {
                sb1[tid] = b1[m * HID + tid];
                sb2[tid] = b2[m * HID + tid];
                sb3[tid] = b3[m * HID + tid];
            }
            if (tid < 16) sb4[tid] = (tid < OUT_DIM) ? b4[m * OUT_DIM + tid] : 0.f;
        }
        __syncthreads();

        // ---- Layer 1: relu(x @ W1 + b1), K=32, both sub-tiles share B ----
#pragma unroll
        for (int t = 0; t < 4; ++t) {
            v16h bf = load_fragB(sW1, IN_DIM, 0, t, r, half);
            float bv = sb1[t * 16 + r];
            v8f acc0 = v8f_splat(bv), acc1 = v8f_splat(bv);
            acc0 = wmma_f16(xa0, bf, acc0);
            acc1 = wmma_f16(xa1, bf, acc1);
            relu_store(hbuf, acc0, 0,  t * 16 + r, half);
            relu_store(hbuf, acc1, 16, t * 16 + r, half);
        }
        v16h a00 = load_fragA(hbuf + r * HID, 0, half);
        v16h a01 = load_fragA(hbuf + r * HID, 32, half);
        v16h a10 = load_fragA(hbuf + (16 + r) * HID, 0, half);
        v16h a11 = load_fragA(hbuf + (16 + r) * HID, 32, half);

        // ---- Layer 2 ----
        layer_hid(sW2, sb2, hbuf, r, half, a00, a01, a10, a11);
        a00 = load_fragA(hbuf + r * HID, 0, half);
        a01 = load_fragA(hbuf + r * HID, 32, half);
        a10 = load_fragA(hbuf + (16 + r) * HID, 0, half);
        a11 = load_fragA(hbuf + (16 + r) * HID, 32, half);

        // ---- Layer 3 ----
        layer_hid(sW3, sb3, hbuf, r, half, a00, a01, a10, a11);
        a00 = load_fragA(hbuf + r * HID, 0, half);
        a01 = load_fragA(hbuf + r * HID, 32, half);
        a10 = load_fragA(hbuf + (16 + r) * HID, 0, half);
        a11 = load_fragA(hbuf + (16 + r) * HID, 32, half);

        // ---- Layer 4: pred = h3 @ W4 + b4 (N padded to 16), accumulate stats ----
        {
            v16h b0 = load_fragB(sW4, HID, 0,  0, r, half);
            v16h b1f = load_fragB(sW4, HID, 32, 0, r, half);
            float bv = sb4[r];
            v8f acc0 = v8f_splat(bv), acc1 = v8f_splat(bv);
            acc0 = wmma_f16(a00, b0, acc0);
            acc1 = wmma_f16(a10, b0, acc1);
            acc0 = wmma_f16(a01, b1f, acc0);
            acc1 = wmma_f16(a11, b1f, acc1);
            psum0 += acc0; psq0 += acc0 * acc0;
            psum1 += acc1; psq1 += acc1 * acc1;
        }
    }

    // ---- Mean / unbiased std over models ----
    if (r < OUT_DIM) {
#pragma unroll
        for (int s = 0; s < 2; ++s) {
            v8f ps = s ? psum1 : psum0;
            v8f pq = s ? psq1  : psq0;
#pragma unroll
            for (int v = 0; v < 8; ++v) {
                int row = rowbase + s * 16 + v + 8 * half;
                float su   = ps[v];
                float mean = su * (1.f / N_MODELS);
                float var  = (pq[v] - su * su * (1.f / N_MODELS)) * (1.f / (N_MODELS - 1));
                float sd   = sqrtf(var > 0.f ? var : 0.f);
                out[(size_t)row * OUT_DIM + r] = mean;
                out[(size_t)batch * OUT_DIM + (size_t)row * OUT_DIM + r] = sd;
            }
        }
    }
}

// Prologue: pack all model weights once (transpose, f16, pad) into d_ws.
__global__ __launch_bounds__(256)
void pack_weights_kernel(const float* __restrict__ W1, const float* __restrict__ b1,
                         const float* __restrict__ W2, const float* __restrict__ b2,
                         const float* __restrict__ W3, const float* __restrict__ b3,
                         const float* __restrict__ W4, const float* __restrict__ b4,
                         unsigned char* __restrict__ blob) {
    const int m = blockIdx.x;
    const int tid = threadIdx.x;
    unsigned char* base = blob + (size_t)m * BLOB_BYTES;
    _Float16* w = (_Float16*)base;

    const float* g = W1 + (size_t)m * IN_DIM * HID;
    for (int e = tid; e < IN_DIM * HID; e += 256) {
        int k = e / HID, n = e % HID;
        w[OFF_W1 + n * IN_DIM + k] = (_Float16)g[e];
    }
    g = W2 + (size_t)m * HID * HID;
    for (int e = tid; e < HID * HID; e += 256) {
        int k = e / HID, n = e % HID;
        w[OFF_W2 + n * HID + k] = (_Float16)g[e];
    }
    g = W3 + (size_t)m * HID * HID;
    for (int e = tid; e < HID * HID; e += 256) {
        int k = e / HID, n = e % HID;
        w[OFF_W3 + n * HID + k] = (_Float16)g[e];
    }
    g = W4 + (size_t)m * HID * OUT_DIM;
    for (int e = tid; e < 16 * HID; e += 256) {
        int n = e >> 6, k = e & 63;
        w[OFF_W4 + e] = (n < OUT_DIM) ? (_Float16)g[k * OUT_DIM + n] : (_Float16)0.f;
    }
    float* bb = (float*)(base + BIAS_OFF_BYTES);
    if (tid < HID) {
        bb[tid]       = b1[m * HID + tid];
        bb[64 + tid]  = b2[m * HID + tid];
        bb[128 + tid] = b3[m * HID + tid];
    }
    if (tid < 16) bb[192 + tid] = (tid < OUT_DIM) ? b4[m * OUT_DIM + tid] : 0.f;
}

extern "C" void kernel_launch(void* const* d_in, const int* in_sizes, int n_in,
                              void* d_out, int out_size, void* d_ws, size_t ws_size,
                              hipStream_t stream) {
    const float* x  = (const float*)d_in[0];
    const float* W1 = (const float*)d_in[1];
    const float* b1 = (const float*)d_in[2];
    const float* W2 = (const float*)d_in[3];
    const float* b2 = (const float*)d_in[4];
    const float* W3 = (const float*)d_in[5];
    const float* b3 = (const float*)d_in[6];
    const float* W4 = (const float*)d_in[7];
    const float* b4 = (const float*)d_in[8];
    float* out = (float*)d_out;

    int batch  = in_sizes[0] / IN_DIM;           // 131072
    int blocks = batch / ROWS_PER_BLOCK;         // 512

    const size_t need = (size_t)N_MODELS * BLOB_BYTES;  // ~365 KB
    if (ws_size >= need) {
        unsigned char* blob = (unsigned char*)d_ws;
        pack_weights_kernel<<<N_MODELS, 256, 0, stream>>>(W1, b1, W2, b2, W3, b3,
                                                          W4, b4, blob);
        ensemble_mlp_kernel<true><<<blocks, 256, 0, stream>>>(
            x, W1, b1, W2, b2, W3, b3, W4, b4, blob, out, batch);
    } else {
        ensemble_mlp_kernel<false><<<blocks, 256, 0, stream>>>(
            x, W1, b1, W2, b2, W3, b3, W4, b4, nullptr, out, batch);
    }
}
